// HashRouterLayer_51719996178902
// MI455X (gfx1250) — matrix operand order repaired and checked by reading, not compiled, not run on hardware
//
#include <hip/hip_runtime.h>
#include <math.h>

typedef _Float16 h8   __attribute__((ext_vector_type(8)));
typedef _Float16 v16h __attribute__((ext_vector_type(16)));
typedef _Float16 h2v  __attribute__((ext_vector_type(2)));
typedef float    v8f  __attribute__((ext_vector_type(8)));

struct LevelParams {
    float scale[16];
    int   res[16];
    int   hsize[16];
    int   off[16];
};

#define WAVES 8
#define PTS_PER_WAVE 16
#define PTS_PER_BLOCK (WAVES * PTS_PER_WAVE)   // 128

// valid for slope in (0,1): v>0 -> v > 0.01v ; v<=0 -> 0.01v >= v
__device__ __forceinline__ float leaky(float v) { return fmaxf(v, 0.01f * v); }

__device__ __forceinline__ v16h pack16(h8 lo, h8 hi) {
    return __builtin_shufflevector(lo, hi, 0,1,2,3,4,5,6,7,8,9,10,11,12,13,14,15);
}

// B operand: wt is f16 [32 cols][32 k] (transposed weights). Lane n holds col n%16
// (+tile offset), halves 0..7 = K koff/2..koff/2+7, 8..15 = K koff/2+16..koff/2+23 (koff = 16 for lanes 16..31).
__device__ __forceinline__ v16h load_b(const _Float16* wt, int col, int koff) {
    const h8* p = (const h8*)(wt + col * 32 + koff / 2);
    return pack16(p[0], p[2]);
}

struct LvlAddr {
    int   idx[8];                 // absolute float2 indices of the 8 corners
    float r0, r1, r2;             // fracs (weights derived at accumulate time)
};

// Compute the 8 absolute corner indices + fracs for one level.
__device__ __forceinline__ LvlAddr make_addrs(float x0, float x1, float x2,
                                              float sc, int rs, int hs, int of) {
    LvlAddr A;
    float p0 = x0 * sc, p1 = x1 * sc, p2 = x2 * sc;
    float f0 = floorf(p0), f1 = floorf(p1), f2 = floorf(p2);
    A.r0 = p0 - f0; A.r1 = p1 - f1; A.r2 = p2 - f2;
    int bx = (int)f0, by = (int)f1, bz = (int)f2;
    int dy = rs, dz = rs * rs;
    int lin000 = bx + by * dy + bz * dz;
    #pragma unroll
    for (int c = 0; c < 8; ++c) {
        const int cx = c & 1, cy = (c >> 1) & 1, cz = (c >> 2) & 1;
        int lin = lin000 + (cx ? 1 : 0) + (cy ? dy : 0) + (cz ? dz : 0);
        // uncapped levels: lin < hs always; capped levels: hs == 2^19 (pow2 mask)
        A.idx[c] = of + ((lin < hs) ? lin : (lin & (hs - 1)));
    }
    return A;
}

__device__ __forceinline__ void accum(const LvlAddr& A, const float2* f,
                                      float& c0, float& c1) {
    const float wx1 = A.r0, wy1 = A.r1, wz1 = A.r2;
    const float wx0 = 1.f - wx1, wy0 = 1.f - wy1, wz0 = 1.f - wz1;
    #pragma unroll
    for (int c = 0; c < 8; ++c) {
        const int cx = c & 1, cy = (c >> 1) & 1, cz = (c >> 2) & 1;
        float w = (cx ? wx1 : wx0) * (cy ? wy1 : wy0) * (cz ? wz1 : wz0);
        c0 = fmaf(w, f[c].x, c0);
        c1 = fmaf(w, f[c].y, c1);
    }
}

__global__ __launch_bounds__(256)
void ngp_fused_kernel(const float* __restrict__ coords,
                      const float* __restrict__ emb,
                      const float* __restrict__ W0, const float* __restrict__ b0,
                      const float* __restrict__ W1, const float* __restrict__ b1,
                      const float* __restrict__ W2, const float* __restrict__ b2,
                      float* __restrict__ out, int npts, LevelParams P)
{
    __shared__ _Float16 encS[WAVES][16][32];   // per-wave 16x32 f16 activation tile
    __shared__ _Float16 w0t[32][32];           // [col][k]
    __shared__ _Float16 w1t[32][32];           // [col][k]
    __shared__ float b0s[32], b1s[32], w2s[32];

    const int tid = threadIdx.x;

    // ---- stage weights: transpose to [col][k] in f16 ----
    for (int j = tid; j < 1024; j += 256) {
        int col = j >> 5, k = j & 31;
        w0t[col][k] = (_Float16)W0[k * 32 + col];
        w1t[col][k] = (_Float16)W1[k * 32 + col];
    }
    if (tid < 32) { b0s[tid] = b0[tid]; b1s[tid] = b1[tid]; w2s[tid] = W2[tid]; }
    __syncthreads();

    const int wave = tid >> 5;
    const int lane = tid & 31;
    const int m    = lane & 15;   // row / point-in-wave / column-in-tile
    const int hi   = lane >> 4;   // half-wave select

    // ---- grid encode: lane handles levels [hi*8, hi*8+8) of point m ----
    int gp = blockIdx.x * PTS_PER_BLOCK + wave * PTS_PER_WAVE + m;
    if (gp >= npts) gp = npts - 1;             // branchless clamp, keeps EXEC full
    const float* cp = coords + gp * 3;
    const float x0 = (cp[0] + 1.f) * 0.5f;
    const float x1 = (cp[1] + 1.f) * 0.5f;
    const float x2 = (cp[2] + 1.f) * 0.5f;

    const float2* embv = (const float2*)emb;

    // Process levels in PAIRS: issue all 16 gathers before consuming any, so the
    // compiler batches global_load_b64s (high MLP against L2) instead of
    // load->wait->fma serialization.
    #pragma unroll
    for (int lp = 0; lp < 8; lp += 2) {
        const int liA = lp, liB = lp + 1;
        const float scA = hi ? P.scale[liA + 8] : P.scale[liA];
        const int   rsA = hi ? P.res[liA + 8]   : P.res[liA];
        const int   hsA = hi ? P.hsize[liA + 8] : P.hsize[liA];
        const int   ofA = hi ? P.off[liA + 8]   : P.off[liA];
        const float scB = hi ? P.scale[liB + 8] : P.scale[liB];
        const int   rsB = hi ? P.res[liB + 8]   : P.res[liB];
        const int   hsB = hi ? P.hsize[liB + 8] : P.hsize[liB];
        const int   ofB = hi ? P.off[liB + 8]   : P.off[liB];

        LvlAddr A = make_addrs(x0, x1, x2, scA, rsA, hsA, ofA);
        LvlAddr B = make_addrs(x0, x1, x2, scB, rsB, hsB, ofB);

        float2 fA[8], fB[8];
        #pragma unroll
        for (int c = 0; c < 8; ++c) fA[c] = embv[A.idx[c]];
        #pragma unroll
        for (int c = 0; c < 8; ++c) fB[c] = embv[B.idx[c]];

        float a0 = 0.f, a1 = 0.f, bb0 = 0.f, bb1 = 0.f;
        accum(A, fA, a0, a1);
        accum(B, fB, bb0, bb1);

        // feature index = level*2 + channel (level-major, matches reference)
        h2v hA; hA.x = (_Float16)a0;  hA.y = (_Float16)a1;
        h2v hB; hB.x = (_Float16)bb0; hB.y = (_Float16)bb1;
        *(h2v*)&encS[wave][m][(hi * 8 + liA) * 2] = hA;  // 4B-aligned ds_store_b32
        *(h2v*)&encS[wave][m][(hi * 8 + liB) * 2] = hB;
    }
    // wave-private LDS region; DS ops from one wave are in order -> no barrier.

    // ---- A operand (16x32 f16): lanes<16 hold K0-7,K16-23; lanes>=16 K8-15,K24-31
    const int kb   = hi * 8;
    const int koff = hi * 16;
    v16h a = pack16(*(const h8*)&encS[wave][m][kb],
                    *(const h8*)&encS[wave][m][kb + 16]);

    // ---- layer 0: enc[16x32] @ W0[32x32], two 16-col tiles ----
    v8f acc0 = {}, acc1 = {};
    {
        v16h bA = load_b(&w0t[0][0], m,      koff);
        v16h bB = load_b(&w0t[0][0], m + 16, koff);
        acc0 = __builtin_amdgcn_wmma_f32_16x16x32_f16(false, a, false, bA, (short)0, acc0, false, false);
        acc1 = __builtin_amdgcn_wmma_f32_16x16x32_f16(false, a, false, bB, (short)0, acc1, false, false);
    }
    {
        const float bb0 = b0s[m], bb1 = b0s[m + 16];
        #pragma unroll
        for (int r = 0; r < 8; ++r) {
            acc0[r] = leaky(acc0[r] + bb0);
            acc1[r] = leaky(acc1[r] + bb1);
        }
    }

    // ---- restage h0 into LDS in row-major f16 (D layout: vgpr r -> row r+8*hi) ----
    #pragma unroll
    for (int r = 0; r < 8; ++r) {
        const int row = r + hi * 8;
        encS[wave][row][m]      = (_Float16)acc0[r];
        encS[wave][row][m + 16] = (_Float16)acc1[r];
    }

    // ---- layer 1 ----
    v16h a1 = pack16(*(const h8*)&encS[wave][m][kb],
                     *(const h8*)&encS[wave][m][kb + 16]);
    v8f d0 = {}, d1 = {};
    {
        v16h bA = load_b(&w1t[0][0], m,      koff);
        v16h bB = load_b(&w1t[0][0], m + 16, koff);
        d0 = __builtin_amdgcn_wmma_f32_16x16x32_f16(false, a1, false, bA, (short)0, d0, false, false);
        d1 = __builtin_amdgcn_wmma_f32_16x16x32_f16(false, a1, false, bB, (short)0, d1, false, false);
    }

    // ---- layer 2 (32 -> 1): fp32 dot + xor-shuffle reduce within 16-lane halves ----
    const float c0b = b1s[m], c1b = b1s[m + 16];
    const float w2a = w2s[m], w2b = w2s[m + 16];
    float partial[8];
    #pragma unroll
    for (int r = 0; r < 8; ++r) {
        float v0 = leaky(d0[r] + c0b);
        float v1 = leaky(d1[r] + c1b);
        partial[r] = v0 * w2a + v1 * w2b;
    }
    #pragma unroll
    for (int mask = 1; mask < 16; mask <<= 1) {
        #pragma unroll
        for (int r = 0; r < 8; ++r)
            partial[r] += __shfl_xor(partial[r], mask, 32);
    }

    const float bias2 = b2[0];
    if (m == 0) {
        const int base = blockIdx.x * PTS_PER_BLOCK + wave * PTS_PER_WAVE + hi * 8;
        #pragma unroll
        for (int r = 0; r < 8; ++r) {
            const int o = base + r;          // row M = r + 8*hi
            if (o < npts) out[o] = partial[r] + bias2;
        }
    }
}

extern "C" void kernel_launch(void* const* d_in, const int* in_sizes, int n_in,
                              void* d_out, int out_size, void* d_ws, size_t ws_size,
                              hipStream_t stream) {
    const float* coords = (const float*)d_in[0];
    const float* emb    = (const float*)d_in[1];
    const float* W0     = (const float*)d_in[2];
    const float* b0     = (const float*)d_in[3];
    const float* W1     = (const float*)d_in[4];
    const float* b1     = (const float*)d_in[5];
    const float* W2     = (const float*)d_in[6];
    const float* b2     = (const float*)d_in[7];
    float* out = (float*)d_out;

    const int npts = in_sizes[0] / 3;

    // torch-ngp gridencoder level math (same double-precision path as numpy)
    LevelParams P;
    const double S = log2(512.0 / 16.0) / 15.0;   // NUM_LEVELS-1 = 15
    long long offs = 0;
    for (int i = 0; i < 16; ++i) {
        double sc = exp2((double)i * S) * 16.0 - 1.0;
        P.scale[i] = (float)sc;
        long long r = (long long)ceil(sc) + 1;
        P.res[i] = (int)r;
        long long sz = r * r * r;
        const long long cap = 1LL << 19;
        if (sz > cap) sz = cap;
        sz = ((sz + 7) / 8) * 8;
        P.hsize[i] = (int)sz;
        P.off[i]   = (int)offs;
        offs += sz;
    }

    const int blocks = (npts + PTS_PER_BLOCK - 1) / PTS_PER_BLOCK;
    ngp_fused_kernel<<<blocks, 256, 0, stream>>>(coords, emb, W0, b0, W1, b1, W2, b2,
                                                 out, npts, P);
}
